// NMN_52321291599932
// MI455X (gfx1250) — matrix operand order repaired and checked by reading, not compiled, not run on hardware
//
#include <hip/hip_runtime.h>
#include <math.h>

// ---------------------------------------------------------------------------
// NMN cell for MI455X (gfx1250, wave32).
// GEMMs: operands pre-converted to f16 (weights stored K-transposed), so each
// wave's inner loop is 10x global_load_b128 + 4x v_wmma_f32_16x16x32_f16 per
// 32-K chunk (16x64 output strip per wave, A fragment reused across 4 WMMAs).
// kb_batch (103 MB) is read exactly twice; all reductions are deterministic.
// ---------------------------------------------------------------------------

#define NBATCH 256
#define PIX    196          // 14*14
#define KBC    512
#define CTRLC  512
#define MEMC   512
#define LSTK   8
#define C2C    1536
#define EPSV   1e-12f

typedef __attribute__((ext_vector_type(16))) _Float16 v16h;
typedef __attribute__((ext_vector_type(8)))  _Float16 v8h;
typedef __attribute__((ext_vector_type(8)))  float    v8f;

// ---- workspace layout (float elements; f16 regions hold 2 halves/slot) ----
constexpr size_t OFF_CF    = 0;                           // c_find     256*512
constexpr size_t OFF_CT    = OFF_CF    + 131072;          // c_transform
constexpr size_t OFF_CDO   = OFF_CT    + 131072;          // c_describeone
constexpr size_t OFF_CDT   = OFF_CDO   + 131072;          // c_describetwo
constexpr size_t OFF_RAWA  = OFF_CDT   + 131072;          // read(stk,ptr)     256*196
constexpr size_t OFF_RAWB  = OFF_RAWA  + 50176;           // read(stk,ptr_bw)
constexpr size_t OFF_SA    = OFF_RAWB  + 50176;           // softmax(rawA)
constexpr size_t OFF_SB    = OFF_SA    + 50176;           // softmax(rawB)
constexpr size_t OFF_PF    = OFF_SB    + 50176;           // move_fw(ptr)      256*8
constexpr size_t OFF_PB    = OFF_PF    + 2048;            // move_bw(ptr)
constexpr size_t OFF_PBF   = OFF_PB    + 2048;            // move_bw(move_fw)
constexpr size_t OFF_KBA   = OFF_PBF   + 2048;            // kb_att from A     256*512
constexpr size_t OFF_KBB   = OFF_KBA   + 131072;          // kb_att from B
constexpr size_t OFF_ATTF  = OFF_KBB   + 131072;          // att_find          256*196
constexpr size_t OFF_ATTS  = OFF_ATTF  + 50176;           // att_scene
constexpr size_t OFF_ATTT  = OFF_ATTS  + 50176;           // att_transform
constexpr size_t OFF_X1    = OFF_ATTT  + 50176;           // concat1           256*1536
constexpr size_t OFF_X2    = OFF_X1    + 393216;          // concat2
constexpr size_t OFF_M1    = OFF_X2    + 393216;          // mem describeone   256*512
constexpr size_t OFF_M2    = OFF_M1    + 131072;          // mem describetwo
// f16 staging (sizes in float units = halves/2)
constexpr size_t OFF_CTRLH = OFF_M2    + 131072;          // ctrl f16   256*512h
constexpr size_t OFF_X1H   = OFF_CTRLH + 65536;           // X1 f16     256*1536h
constexpr size_t OFF_X2H   = OFF_X1H   + 196608;          // X2 f16
constexpr size_t OFF_WTF   = OFF_X2H   + 196608;          // find_ci_w^T   f16 512x512h
constexpr size_t OFF_WTT   = OFF_WTF   + 131072;          // transform_ci_w^T
constexpr size_t OFF_WTDO  = OFF_WTT   + 131072;          // describeone_ci_w^T
constexpr size_t OFF_WTDT  = OFF_WTDO  + 131072;          // describetwo_ci_w^T
constexpr size_t OFF_WTM1  = OFF_WTDT  + 131072;          // describeone_mem_w^T 512x1536h
constexpr size_t OFF_WTM2  = OFF_WTM1  + 393216;          // describetwo_mem_w^T

// ---------------------------------------------------------------------------
// f32 -> f16 row convert (elementwise)
// ---------------------------------------------------------------------------
__global__ __launch_bounds__(256) void k_f32_to_f16(
    const float* __restrict__ in, _Float16* __restrict__ out, int count)
{
  const int i = blockIdx.x * blockDim.x + threadIdx.x;
  if (i < count) out[i] = (_Float16)in[i];
}

// ---------------------------------------------------------------------------
// f32 [K x N] -> f16 [N x K] tiled transpose-convert through LDS (32x32 tile).
// ---------------------------------------------------------------------------
__global__ __launch_bounds__(256) void k_transpose_f16(
    const float* __restrict__ W, _Float16* __restrict__ WT, int K, int N)
{
  __shared__ float tile[32][33];
  const int tilesN = N >> 5;
  const int tk = blockIdx.x / tilesN;
  const int tn = blockIdx.x % tilesN;
  const int tx = threadIdx.x & 31;
  const int ty = threadIdx.x >> 5;   // 0..7
  const int k0 = tk << 5, n0 = tn << 5;
#pragma unroll
  for (int r = ty; r < 32; r += 8)
    tile[r][tx] = W[(size_t)(k0 + r) * N + (n0 + tx)];
  __syncthreads();
#pragma unroll
  for (int r = ty; r < 32; r += 8)
    WT[(size_t)(n0 + r) * K + (k0 + tx)] = (_Float16)tile[tx][r];
}

// ---------------------------------------------------------------------------
// WMMA GEMM: C[M,N] = A[M,K](f16,row-major) * WT[N,K](f16, K-transposed) + bias
// One wave computes a 16x64 output strip (4 N-tiles, 4 accumulators).
// Fragment layouts used here (contiguous k = 16h + s, same for A and B):
//   A 16x32 f16 : lane row = lane&15; K run kA..kA+15,
//                 kA = (lane<16 ? 0 : 16)          -> two v8h (b128) loads
//   B 32x16 f16 : lane col = lane&15; K run kB..kB+15, kB = (lane<16 ? 0 :16)
//                 contiguous in WT                 -> two v8h (b128) loads
//   D 16x16 f32 : col = lane&15, row = (lane<16?0:8) + v
// Requires: M%16==0, N%64==0, K%32==0.
// ---------------------------------------------------------------------------
__global__ __launch_bounds__(256) void wmma_gemm_f16t(
    const _Float16* __restrict__ A, const _Float16* __restrict__ WT,
    const float* __restrict__ bias, float* __restrict__ C,
    int M, int N, int K)
{
  const int lane = threadIdx.x & 31;
  const int wave = threadIdx.x >> 5;
  const int ngrp = N >> 6;                 // groups of 4 N-tiles
  const int work = (M >> 4) * ngrp;
  const int wi   = blockIdx.x * 8 + wave;
  if (wi >= work) return;
  const int tm  = wi / ngrp;
  const int tng = wi % ngrp;

  const int rowA  = (tm << 4) + (lane & 15);
  const int colB0 = (tng << 6) + (lane & 15);
  const int kA    = (lane < 16) ? 0 : 16;    // A-fragment K base shift
  const int kB    = (lane < 16) ? 0 : 16;    // B-fragment K base shift

  const _Float16* Ap = A + (size_t)rowA * K + kA;
  const _Float16* Wp0 = WT + (size_t)(colB0     ) * K + kB;
  const _Float16* Wp1 = WT + (size_t)(colB0 + 16) * K + kB;
  const _Float16* Wp2 = WT + (size_t)(colB0 + 32) * K + kB;
  const _Float16* Wp3 = WT + (size_t)(colB0 + 48) * K + kB;

  v8f acc[4];
#pragma unroll
  for (int t = 0; t < 4; ++t) acc[t] = (v8f){0.f,0.f,0.f,0.f,0.f,0.f,0.f,0.f};

  for (int kk = 0; kk < K; kk += 32) {
    const v8h a0 = *(const v8h*)(Ap + kk);
    const v8h a1 = *(const v8h*)(Ap + kk + 8);
    const v16h a = __builtin_shufflevector(a0, a1, 0, 1, 2, 3, 4, 5, 6, 7,
                                                   8, 9, 10, 11, 12, 13, 14, 15);
    const v8h b00 = *(const v8h*)(Wp0 + kk);
    const v8h b01 = *(const v8h*)(Wp0 + kk + 8);
    const v16h b0 = __builtin_shufflevector(b00, b01, 0, 1, 2, 3, 4, 5, 6, 7,
                                                      8, 9, 10, 11, 12, 13, 14, 15);
    acc[0] = __builtin_amdgcn_wmma_f32_16x16x32_f16(
        false, a, false, b0, (short)0, acc[0], false, false);

    const v8h b10 = *(const v8h*)(Wp1 + kk);
    const v8h b11 = *(const v8h*)(Wp1 + kk + 8);
    const v16h b1 = __builtin_shufflevector(b10, b11, 0, 1, 2, 3, 4, 5, 6, 7,
                                                      8, 9, 10, 11, 12, 13, 14, 15);
    acc[1] = __builtin_amdgcn_wmma_f32_16x16x32_f16(
        false, a, false, b1, (short)0, acc[1], false, false);

    const v8h b20 = *(const v8h*)(Wp2 + kk);
    const v8h b21 = *(const v8h*)(Wp2 + kk + 8);
    const v16h b2 = __builtin_shufflevector(b20, b21, 0, 1, 2, 3, 4, 5, 6, 7,
                                                      8, 9, 10, 11, 12, 13, 14, 15);
    acc[2] = __builtin_amdgcn_wmma_f32_16x16x32_f16(
        false, a, false, b2, (short)0, acc[2], false, false);

    const v8h b30 = *(const v8h*)(Wp3 + kk);
    const v8h b31 = *(const v8h*)(Wp3 + kk + 8);
    const v16h b3 = __builtin_shufflevector(b30, b31, 0, 1, 2, 3, 4, 5, 6, 7,
                                                      8, 9, 10, 11, 12, 13, 14, 15);
    acc[3] = __builtin_amdgcn_wmma_f32_16x16x32_f16(
        false, a, false, b3, (short)0, acc[3], false, false);
  }

  const int rbase = (tm << 4) + ((lane < 16) ? 0 : 8);
#pragma unroll
  for (int t = 0; t < 4; ++t) {
    const int col = colB0 + 16 * t;
    const float bv = bias[col];
#pragma unroll
    for (int v = 0; v < 8; ++v)
      C[(size_t)(rbase + v) * N + col] = acc[t][v] + bv;
  }
}

// ---- deterministic block reductions (256 threads) -------------------------
__device__ __forceinline__ float block_max256(float v, float* red) {
  int tid = threadIdx.x;
  red[tid] = v; __syncthreads();
  for (int s = 128; s > 0; s >>= 1) {
    if (tid < s) red[tid] = fmaxf(red[tid], red[tid + s]);
    __syncthreads();
  }
  float r = red[0]; __syncthreads();
  return r;
}
__device__ __forceinline__ float block_sum256(float v, float* red) {
  int tid = threadIdx.x;
  red[tid] = v; __syncthreads();
  for (int s = 128; s > 0; s >>= 1) {
    if (tid < s) red[tid] += red[tid + s];
    __syncthreads();
  }
  float r = red[0]; __syncthreads();
  return r;
}

// ---------------------------------------------------------------------------
// K1: pointer moves + stack reads + spatial softmaxes. One block per n.
// ---------------------------------------------------------------------------
__global__ __launch_bounds__(256) void k_stack_read(
    const float* __restrict__ stk, const float* __restrict__ ptr,
    float* __restrict__ rawA, float* __restrict__ rawB,
    float* __restrict__ softA, float* __restrict__ softB,
    float* __restrict__ ptrF, float* __restrict__ ptrB, float* __restrict__ ptrBF)
{
  const int n = blockIdx.x, tid = threadIdx.x;
  __shared__ float sp[8], spb[8];
  __shared__ float red[256];
  if (tid == 0) {
    float p[8], f[8], b[8], bf[8];
#pragma unroll
    for (int l = 0; l < 8; ++l) p[l] = ptr[n * 8 + l];
    // move_fw
#pragma unroll
    for (int l = 0; l < 8; ++l) f[l] = (l == 0) ? 0.f : p[l - 1];
    f[7] += p[7];
    // move_bw
#pragma unroll
    for (int l = 0; l < 8; ++l) b[l] = (l == 7) ? 0.f : p[l + 1];
    b[0] += p[0];
    // move_bw(move_fw(p))
#pragma unroll
    for (int l = 0; l < 8; ++l) bf[l] = (l == 7) ? 0.f : f[l + 1];
    bf[0] += f[0];
#pragma unroll
    for (int l = 0; l < 8; ++l) {
      sp[l] = p[l]; spb[l] = b[l];
      ptrF[n * 8 + l] = f[l]; ptrB[n * 8 + l] = b[l]; ptrBF[n * 8 + l] = bf[l];
    }
  }
  __syncthreads();

  const bool valid = tid < PIX;
  float a = 0.f, bb = 0.f;
  if (valid) {
    const float* s = stk + ((size_t)n * PIX + tid) * LSTK;
#pragma unroll
    for (int l = 0; l < 8; ++l) { float v = s[l]; a += v * sp[l]; bb += v * spb[l]; }
    rawA[n * PIX + tid] = a;
    rawB[n * PIX + tid] = bb;
  }
  // softmax A over 196 pixels
  float m = block_max256(valid ? a : -3.402823466e38f, red);
  float e = valid ? __expf(a - m) : 0.f;
  float s = block_sum256(e, red);
  if (valid) softA[n * PIX + tid] = e / s;
  // softmax B
  m = block_max256(valid ? bb : -3.402823466e38f, red);
  e = valid ? __expf(bb - m) : 0.f;
  s = block_sum256(e, red);
  if (valid) softB[n * PIX + tid] = e / s;
}

// ---------------------------------------------------------------------------
// K2: single pass over kb:  att_find, att_scene, kbA, kbB.  One block per n.
// Waves own pixels; lanes own channel slices (ch = lane + 32j).
// ---------------------------------------------------------------------------
__global__ __launch_bounds__(256) void k_kb_pass1(
    const float* __restrict__ kb, const float* __restrict__ cF,
    const float* __restrict__ wF, const float* __restrict__ bF,
    const float* __restrict__ wS, const float* __restrict__ bS,
    const float* __restrict__ softA, const float* __restrict__ softB,
    float* __restrict__ attF, float* __restrict__ attS,
    float* __restrict__ kbA, float* __restrict__ kbB)
{
  const int n = blockIdx.x, tid = threadIdx.x;
  __shared__ float s_cf[512], s_wf[512], s_ws[512];
  __shared__ float s_pA[8 * 512], s_pB[8 * 512];   // per-wave partials (deterministic)
  for (int i = tid; i < 512; i += 256) {
    s_cf[i] = cF[(size_t)n * 512 + i]; s_wf[i] = wF[i]; s_ws[i] = wS[i];
  }
  __syncthreads();

  const int wid = tid >> 5, lane = tid & 31;
  const float fb = bF[0], sb = bS[0];
  float accA[16], accB[16];
#pragma unroll
  for (int j = 0; j < 16; ++j) { accA[j] = 0.f; accB[j] = 0.f; }

  for (int p = wid; p < PIX; p += 8) {
    const float sA = softA[n * PIX + p], sB = softB[n * PIX + p];
    const float* kp = kb + ((size_t)n * PIX + p) * 512;
    float df = 0.f, sf = 0.f, dsv = 0.f, ssv = 0.f;
#pragma unroll
    for (int j = 0; j < 16; ++j) {
      const int c = lane + 32 * j;
      const float kv = kp[c];
      const float xf = kv * s_cf[c];
      df  += xf * s_wf[c];  sf  += xf * xf;
      dsv += kv * s_ws[c];  ssv += kv * kv;
      accA[j] += kv * sA;   accB[j] += kv * sB;
    }
#pragma unroll
    for (int msk = 16; msk >= 1; msk >>= 1) {
      df  += __shfl_xor(df,  msk, 32);
      sf  += __shfl_xor(sf,  msk, 32);
      dsv += __shfl_xor(dsv, msk, 32);
      ssv += __shfl_xor(ssv, msk, 32);
    }
    if (lane == 0) {
      attF[n * PIX + p] = df  / fmaxf(sqrtf(sf),  EPSV) + fb;
      attS[n * PIX + p] = dsv / fmaxf(sqrtf(ssv), EPSV) + sb;
    }
  }
#pragma unroll
  for (int j = 0; j < 16; ++j) {
    s_pA[wid * 512 + lane + 32 * j] = accA[j];
    s_pB[wid * 512 + lane + 32 * j] = accB[j];
  }
  __syncthreads();
  for (int c = tid; c < 512; c += 256) {
    float a = 0.f, b = 0.f;
#pragma unroll
    for (int w = 0; w < 8; ++w) { a += s_pA[w * 512 + c]; b += s_pB[w * 512 + c]; }
    kbA[(size_t)n * 512 + c] = a;
    kbB[(size_t)n * 512 + c] = b;
  }
}

// ---------------------------------------------------------------------------
// K3: second kb pass -> att_transform (needs kbA). One block per n.
// ---------------------------------------------------------------------------
__global__ __launch_bounds__(256) void k_kb_transform(
    const float* __restrict__ kb, const float* __restrict__ cT,
    const float* __restrict__ wT, const float* __restrict__ bT,
    const float* __restrict__ kbA, float* __restrict__ attT)
{
  const int n = blockIdx.x, tid = threadIdx.x;
  __shared__ float s_c[512], s_w[512], s_a[512];
  for (int i = tid; i < 512; i += 256) {
    s_c[i] = cT[(size_t)n * 512 + i]; s_w[i] = wT[i]; s_a[i] = kbA[(size_t)n * 512 + i];
  }
  __syncthreads();
  const int wid = tid >> 5, lane = tid & 31;
  const float tb = bT[0];
  for (int p = wid; p < PIX; p += 8) {
    const float* kp = kb + ((size_t)n * PIX + p) * 512;
    float d = 0.f, s = 0.f;
#pragma unroll
    for (int j = 0; j < 16; ++j) {
      const int c = lane + 32 * j;
      const float x = kp[c] * s_c[c] * s_a[c];
      d += x * s_w[c]; s += x * x;
    }
#pragma unroll
    for (int msk = 16; msk >= 1; msk >>= 1) {
      d += __shfl_xor(d, msk, 32);
      s += __shfl_xor(s, msk, 32);
    }
    if (lane == 0) attT[n * PIX + p] = d / fmaxf(sqrtf(s), EPSV) + tb;
  }
}

// ---------------------------------------------------------------------------
// K4: build concat inputs X1=[ctrl, l2norm(cDO*kbA), kbA], X2=[ctrl, l2norm(cDT*kbA*kbB), kbB]
// ---------------------------------------------------------------------------
__global__ __launch_bounds__(256) void k_build_concat(
    const float* __restrict__ ctrl, const float* __restrict__ cDO,
    const float* __restrict__ cDT, const float* __restrict__ kbA,
    const float* __restrict__ kbB, float* __restrict__ X1, float* __restrict__ X2)
{
  const int n = blockIdx.x, tid = threadIdx.x;
  __shared__ float red[256];
  const size_t base = (size_t)n * 512;

  const float a0 = kbA[base + tid],       a1 = kbA[base + tid + 256];
  const float b0 = kbB[base + tid],       b1 = kbB[base + tid + 256];
  const float e10 = cDO[base + tid] * a0, e11 = cDO[base + tid + 256] * a1;
  const float e20 = cDT[base + tid] * a0 * b0, e21 = cDT[base + tid + 256] * a1 * b1;

  const float n1 = fmaxf(sqrtf(block_sum256(e10 * e10 + e11 * e11, red)), EPSV);
  const float n2 = fmaxf(sqrtf(block_sum256(e20 * e20 + e21 * e21, red)), EPSV);

  const size_t xb = (size_t)n * C2C;
  X1[xb + tid]        = ctrl[base + tid];       X1[xb + tid + 256]        = ctrl[base + tid + 256];
  X1[xb + 512 + tid]  = e10 / n1;               X1[xb + 512 + tid + 256]  = e11 / n1;
  X1[xb + 1024 + tid] = a0;                     X1[xb + 1024 + tid + 256] = a1;
  X2[xb + tid]        = ctrl[base + tid];       X2[xb + tid + 256]        = ctrl[base + tid + 256];
  X2[xb + 512 + tid]  = e20 / n2;               X2[xb + 512 + tid + 256]  = e21 / n2;
  X2[xb + 1024 + tid] = b0;                     X2[xb + 1024 + tid + 256] = b1;
}

// ---------------------------------------------------------------------------
// K5: elementwise 9-module combine for the attention stack. One thread per (n,p).
// ---------------------------------------------------------------------------
__global__ __launch_bounds__(256) void k_combine_att(
    const float* __restrict__ stk, const float* __restrict__ prob,
    const float* __restrict__ ptrP, const float* __restrict__ ptrF,
    const float* __restrict__ ptrB, const float* __restrict__ ptrBF,
    const float* __restrict__ rawA, const float* __restrict__ rawB,
    const float* __restrict__ attF, const float* __restrict__ attT,
    const float* __restrict__ attS, float* __restrict__ out)
{
  const int idx = blockIdx.x * blockDim.x + threadIdx.x;
  if (idx >= NBATCH * PIX) return;
  const int n = idx / PIX;

  float pr[9];
#pragma unroll
  for (int m = 0; m < 9; ++m) pr[m] = prob[n * 9 + m];
  float pt[8], pf[8], pb[8], pbf[8], st[8];
#pragma unroll
  for (int l = 0; l < 8; ++l) {
    pt[l] = ptrP[n * 8 + l]; pf[l] = ptrF[n * 8 + l];
    pb[l] = ptrB[n * 8 + l]; pbf[l] = ptrBF[n * 8 + l];
    st[l] = stk[(size_t)idx * LSTK + l];
  }
  const float aF = attF[idx], aT = attT[idx], aS = attS[idx];
  const float A = rawA[idx], B = rawB[idx];          // att2 = A, att1 = B
  const float mn = fminf(B, A), mx = fmaxf(B, A);

  // Filter = Find then And
  float stk1[8], a1 = 0.f, a2 = 0.f;
#pragma unroll
  for (int l = 0; l < 8; ++l) {
    stk1[l] = aF * pf[l] + st[l] * (1.f - pf[l]);
    a2 += stk1[l] * pf[l];
    a1 += stk1[l] * pbf[l];
  }
  const float mnf = fminf(a1, a2);
  const float pAnd = pr[4] + pr[8];

#pragma unroll
  for (int l = 0; l < 8; ++l) {
    float o = (pr[0] + pr[7]) * st[l]
            + pr[1] * (aF  * pf[l]  + st[l]   * (1.f - pf[l]))
            + pr[2] * (aT  * pt[l]  + st[l]   * (1.f - pt[l]))
            + pr[3] * (mnf * pbf[l] + stk1[l] * (1.f - pbf[l]))
            + pAnd  * (mn  * pb[l]  + st[l]   * (1.f - pb[l]))
            + pr[5] * (mx  * pb[l]  + st[l]   * (1.f - pb[l]))
            + pr[6] * (aS  * pf[l]  + st[l]   * (1.f - pf[l]));
    out[(size_t)idx * LSTK + l] = o;
  }
}

// ---------------------------------------------------------------------------
// K6: pointer average + sharpen (softmax) and mem average. One block per n.
// ---------------------------------------------------------------------------
__global__ __launch_bounds__(256) void k_combine_ptr_mem(
    const float* __restrict__ prob, const float* __restrict__ ptrP,
    const float* __restrict__ ptrF, const float* __restrict__ ptrB,
    const float* __restrict__ ptrBF, const float* __restrict__ memPrev,
    const float* __restrict__ mem1, const float* __restrict__ mem2,
    float* __restrict__ outPtr, float* __restrict__ outMem)
{
  const int n = blockIdx.x, tid = threadIdx.x;
  if (tid == 0) {
    float pr[9];
#pragma unroll
    for (int m = 0; m < 9; ++m) pr[m] = prob[n * 9 + m];
    const float w_t  = pr[0] + pr[2] + pr[7];
    const float w_f  = pr[1] + pr[6];
    const float w_b  = pr[4] + pr[5] + pr[8];
    const float w_bf = pr[3];
    float pv[8], mxv = -3.402823466e38f;
#pragma unroll
    for (int l = 0; l < 8; ++l) {
      pv[l] = w_t * ptrP[n * 8 + l] + w_f * ptrF[n * 8 + l]
            + w_b * ptrB[n * 8 + l] + w_bf * ptrBF[n * 8 + l];
      mxv = fmaxf(mxv, pv[l]);
    }
    float s = 0.f;
#pragma unroll
    for (int l = 0; l < 8; ++l) { pv[l] = __expf(pv[l] - mxv); s += pv[l]; }
#pragma unroll
    for (int l = 0; l < 8; ++l) outPtr[n * 8 + l] = pv[l] / s;
  }
  const float p0 = prob[n * 9 + 0], p7 = prob[n * 9 + 7], p8 = prob[n * 9 + 8];
  for (int c = tid; c < MEMC; c += 256) {
    const size_t i = (size_t)n * MEMC + c;
    outMem[i] = p0 * memPrev[i] + p7 * mem1[i] + p8 * mem2[i];
  }
}

// ---------------------------------------------------------------------------
extern "C" void kernel_launch(void* const* d_in, const int* in_sizes, int n_in,
                              void* d_out, int out_size, void* d_ws, size_t ws_size,
                              hipStream_t stream) {
  (void)in_sizes; (void)n_in; (void)out_size; (void)ws_size;

  const float* ctrl   = (const float*)d_in[0];
  const float* kb     = (const float*)d_in[1];
  const float* prob   = (const float*)d_in[2];
  const float* memP   = (const float*)d_in[3];
  const float* stk    = (const float*)d_in[4];
  const float* ptrP   = (const float*)d_in[5];
  const float* fciw   = (const float*)d_in[6];
  const float* fcib   = (const float*)d_in[7];
  const float* fcw    = (const float*)d_in[8];
  const float* fcb    = (const float*)d_in[9];
  const float* tciw   = (const float*)d_in[10];
  const float* tcib   = (const float*)d_in[11];
  const float* tcw    = (const float*)d_in[12];
  const float* tcb    = (const float*)d_in[13];
  const float* scw    = (const float*)d_in[14];
  const float* scb    = (const float*)d_in[15];
  const float* dociw  = (const float*)d_in[16];
  const float* docib  = (const float*)d_in[17];
  const float* domw   = (const float*)d_in[18];
  const float* domb   = (const float*)d_in[19];
  const float* dtciw  = (const float*)d_in[20];
  const float* dtcib  = (const float*)d_in[21];
  const float* dtmw   = (const float*)d_in[22];
  const float* dtmb   = (const float*)d_in[23];

  float* w = (float*)d_ws;
  float* cF   = w + OFF_CF;   float* cT   = w + OFF_CT;
  float* cDO  = w + OFF_CDO;  float* cDT  = w + OFF_CDT;
  float* rawA = w + OFF_RAWA; float* rawB = w + OFF_RAWB;
  float* sA   = w + OFF_SA;   float* sB   = w + OFF_SB;
  float* pF   = w + OFF_PF;   float* pB   = w + OFF_PB;   float* pBF = w + OFF_PBF;
  float* kbA  = w + OFF_KBA;  float* kbB  = w + OFF_KBB;
  float* atF  = w + OFF_ATTF; float* atS  = w + OFF_ATTS; float* atT = w + OFF_ATTT;
  float* X1   = w + OFF_X1;   float* X2   = w + OFF_X2;
  float* m1   = w + OFF_M1;   float* m2   = w + OFF_M2;

  _Float16* ctrlH = (_Float16*)(w + OFF_CTRLH);
  _Float16* X1H   = (_Float16*)(w + OFF_X1H);
  _Float16* X2H   = (_Float16*)(w + OFF_X2H);
  _Float16* wtF   = (_Float16*)(w + OFF_WTF);
  _Float16* wtT   = (_Float16*)(w + OFF_WTT);
  _Float16* wtDO  = (_Float16*)(w + OFF_WTDO);
  _Float16* wtDT  = (_Float16*)(w + OFF_WTDT);
  _Float16* wtM1  = (_Float16*)(w + OFF_WTM1);
  _Float16* wtM2  = (_Float16*)(w + OFF_WTM2);

  float* out     = (float*)d_out;
  float* outAtt  = out;                                  // 256*196*8
  float* outPtr  = out + (size_t)NBATCH * PIX * LSTK;    // 256*8
  float* outMem  = outPtr + (size_t)NBATCH * LSTK;       // 256*512

  // ---- stage f16 operands --------------------------------------------------
  k_f32_to_f16<<<(NBATCH * CTRLC + 255) / 256, 256, 0, stream>>>(ctrl, ctrlH,
                                                                 NBATCH * CTRLC);
  // ci weights: [512 x 512] -> 256 tiles;   mem weights: [1536 x 512] -> 768
  k_transpose_f16<<<256, 256, 0, stream>>>(fciw,  wtF,  CTRLC, KBC);
  k_transpose_f16<<<256, 256, 0, stream>>>(tciw,  wtT,  CTRLC, KBC);
  k_transpose_f16<<<256, 256, 0, stream>>>(dociw, wtDO, CTRLC, KBC);
  k_transpose_f16<<<256, 256, 0, stream>>>(dtciw, wtDT, CTRLC, KBC);
  k_transpose_f16<<<768, 256, 0, stream>>>(domw,  wtM1, C2C, MEMC);
  k_transpose_f16<<<768, 256, 0, stream>>>(dtmw,  wtM2, C2C, MEMC);

  // ---- ci projections: (256/16)*(512/64) = 128 waves -> 16 blocks ---------
  wmma_gemm_f16t<<<16, 256, 0, stream>>>(ctrlH, wtF,  fcib,  cF,  NBATCH, KBC, CTRLC);
  wmma_gemm_f16t<<<16, 256, 0, stream>>>(ctrlH, wtT,  tcib,  cT,  NBATCH, KBC, CTRLC);
  wmma_gemm_f16t<<<16, 256, 0, stream>>>(ctrlH, wtDO, docib, cDO, NBATCH, KBC, CTRLC);
  wmma_gemm_f16t<<<16, 256, 0, stream>>>(ctrlH, wtDT, dtcib, cDT, NBATCH, KBC, CTRLC);

  k_stack_read<<<NBATCH, 256, 0, stream>>>(stk, ptrP, rawA, rawB, sA, sB, pF, pB, pBF);

  k_kb_pass1<<<NBATCH, 256, 0, stream>>>(kb, cF, fcw, fcb, scw, scb, sA, sB,
                                         atF, atS, kbA, kbB);

  k_kb_transform<<<NBATCH, 256, 0, stream>>>(kb, cT, tcw, tcb, kbA, atT);

  k_build_concat<<<NBATCH, 256, 0, stream>>>(ctrl, cDO, cDT, kbA, kbB, X1, X2);

  // ---- mem heads -----------------------------------------------------------
  k_f32_to_f16<<<(NBATCH * C2C + 255) / 256, 256, 0, stream>>>(X1, X1H, NBATCH * C2C);
  k_f32_to_f16<<<(NBATCH * C2C + 255) / 256, 256, 0, stream>>>(X2, X2H, NBATCH * C2C);
  wmma_gemm_f16t<<<16, 256, 0, stream>>>(X1H, wtM1, domb, m1, NBATCH, MEMC, C2C);
  wmma_gemm_f16t<<<16, 256, 0, stream>>>(X2H, wtM2, dtmb, m2, NBATCH, MEMC, C2C);

  k_combine_att<<<(NBATCH * PIX + 255) / 256, 256, 0, stream>>>(
      stk, prob, ptrP, pF, pB, pBF, rawA, rawB, atF, atT, atS, outAtt);

  k_combine_ptr_mem<<<NBATCH, 256, 0, stream>>>(prob, ptrP, pF, pB, pBF,
                                                memP, m1, m2, outPtr, outMem);
}
